// STABLESSM_75565654606233
// MI455X (gfx1250) — matrix operand order repaired and checked by reading, not compiled, not run on hardware
//
#include <hip/hip_runtime.h>
#include <hip/hip_bf16.h>
#include <math.h>

// Problem dims (fixed by reference)
#define Hd   1024
#define N2d  32
#define Bd   8
#define Ld   4096
#define Md   (Bd * Ld)       // 32768 columns of the GEMM

typedef __attribute__((ext_vector_type(16))) __bf16 v16bf;
typedef __attribute__((ext_vector_type(8)))  float  v8f;

union Frag { uint4 q[2]; v16bf v; };

// Async global->LDS path (CDNA5 GLOBAL_LOAD_ASYNC_TO_LDS_B128), guarded so the
// kernel still compiles on toolchains without the builtin.
#if defined(__has_builtin)
#if __has_builtin(__builtin_amdgcn_global_load_async_to_lds_b128)
#define USE_ASYNC_LDS 1
#endif
#endif

#ifdef USE_ASYNC_LDS
typedef int v4i_ __attribute__((vector_size(16)));
typedef __attribute__((address_space(1))) v4i_* as1_v4i;
typedef __attribute__((address_space(3))) v4i_* as3_v4i;
#endif

__device__ __forceinline__ void cp16(__bf16* lds, const __bf16* glob) {
#ifdef USE_ASYNC_LDS
    __builtin_amdgcn_global_load_async_to_lds_b128(
        (as1_v4i)(v4i_*)(void*)const_cast<__bf16*>(glob),
        (as3_v4i)(v4i_*)(void*)lds,
        0, 0);
#else
    *reinterpret_cast<uint4*>(lds) = *reinterpret_cast<const uint4*>(glob);
#endif
}

__device__ __forceinline__ void async_fence() {
#ifdef USE_ASYNC_LDS
#if __has_builtin(__builtin_amdgcn_s_wait_asynccnt)
    __builtin_amdgcn_s_wait_asynccnt(0);
#else
    asm volatile("s_wait_asynccnt 0" ::: "memory");
#endif
#endif
}

// ---------------------------------------------------------------------------
// Kernel 1: per-(h,n) SSM constants.
//   w  = exp(dt*A),  c2 = 2 * C * (exp(dt*A)-1)/A
// ---------------------------------------------------------------------------
__global__ void ssm_prep_consts(const float* __restrict__ log_dt,
                                const float* __restrict__ C_real,
                                const float* __restrict__ log_A_real,
                                const float* __restrict__ A_imag,
                                float* __restrict__ cst) {
    int i = blockIdx.x * blockDim.x + threadIdx.x;   // 0 .. H*N2-1
    int h = i >> 5;
    float dt = expf(log_dt[h]);
    float ar = -expf(log_A_real[i]);
    float ai = A_imag[i];
    float dre = ar * dt, dim = ai * dt;
    float e  = expf(dre);
    float wr = e * cosf(dim);
    float wi = e * sinf(dim);
    float nr = wr - 1.0f, ni = wi;                   // (w-1)/A via conj(A)/|A|^2
    float den = ar * ar + ai * ai;
    float ir = ar / den, ii = -ai / den;
    float fr = nr * ir - ni * ii;
    float fi = nr * ii + ni * ir;
    float cRe = C_real[2 * i], cIm = C_real[2 * i + 1];
    float c2r = 2.0f * (cRe * fr - cIm * fi);
    float c2i = 2.0f * (cRe * fi + cIm * fr);
    *reinterpret_cast<float4*>(cst + 4 * i) = make_float4(wr, wi, c2r, c2i);
}

// ---------------------------------------------------------------------------
// Kernel 2: W_out f32 -> bf16
// ---------------------------------------------------------------------------
__global__ void w_to_bf16(const float* __restrict__ W, __hip_bfloat16* __restrict__ Wb) {
    int i = blockIdx.x * blockDim.x + threadIdx.x;   // 0 .. 2H*H-1
    Wb[i] = __float2bfloat16(W[i]);
}

// ---------------------------------------------------------------------------
// Kernel 3: diagonal SSM recurrence (replaces FFT conv) + D-skip + exact GELU.
// 4 modes per lane, 8 lanes per channel, 4 channels per wave (2048 waves).
// ---------------------------------------------------------------------------
__global__ __launch_bounds__(256)
void ssm_scan(const float* __restrict__ u,
              const float* __restrict__ cst,
              const float* __restrict__ Dv,
              __hip_bfloat16* __restrict__ Yt) {
    int tid  = blockIdx.x * blockDim.x + threadIdx.x;
    int wid  = tid >> 5;                 // 0 .. B*H/4 - 1
    int lane = threadIdx.x & 31;
    int g = lane >> 3;                   // channel slot in wave (0..3)
    int j = lane & 7;                    // mode-group (0..7), modes 4j..4j+3
    int chan = wid * 4 + g;              // global b*H + h (same b across wave)
    int b = chan >> 10;
    int h = chan & (Hd - 1);

    const float4* c4 = reinterpret_cast<const float4*>(cst) + (h * N2d + j * 4);
    float wr[4], wi[4], cr[4], ci[4];
    #pragma unroll
    for (int k = 0; k < 4; ++k) {
        float4 c = c4[k];
        wr[k] = c.x; wi[k] = c.y; cr[k] = c.z; ci[k] = c.w;
    }
    float d = Dv[h];
    const float* up = u + (size_t)chan * Ld;
    __hip_bfloat16* yp = Yt + (size_t)b * Ld * Hd + h;

    float sr[4] = {0.f, 0.f, 0.f, 0.f};
    float si[4] = {0.f, 0.f, 0.f, 0.f};
    #pragma unroll 2
    for (int l = 0; l < Ld; ++l) {
        float uv = __builtin_nontemporal_load(up + l);
        float acc = 0.0f;
        #pragma unroll
        for (int k = 0; k < 4; ++k) {
            float t   = fmaf(-wi[k], si[k], uv);
            float nsr = fmaf(wr[k], sr[k], t);
            float nsi = fmaf(wr[k], si[k], wi[k] * sr[k]);
            sr[k] = nsr; si[k] = nsi;
            acc = fmaf(cr[k], nsr, acc);
            acc = fmaf(-ci[k], nsi, acc);
        }
        acc += __shfl_xor(acc, 1, 32);
        acc += __shfl_xor(acc, 2, 32);
        acc += __shfl_xor(acc, 4, 32);
        float y = fmaf(d, uv, acc);
        float ge = 0.5f * y * (1.0f + erff(y * 0.70710678118654752f));
        if (j == 0) yp[(size_t)l * Hd] = __float2bfloat16(ge);
    }
}

// ---------------------------------------------------------------------------
// Kernel 4: z = W(2048x1024) @ Y(1024x32768) + b, GLU: out = a * sigmoid(g).
// Double-buffered LDS staging, ONE barrier per K=64 stage, zero staging
// registers: stage kt fills nbuf (== the buffer consumed at kt-1, protected
// by the kt-1 barrier) before computing from cbuf. Fill is an async
// global->LDS DMA when available (ASYNCcnt + s_wait_asynccnt), else a short
// load/ds_store pair. XOR chunk swizzle keeps ds_load_b128 frags conflict-free.
// ---------------------------------------------------------------------------
__device__ __forceinline__ int swz(int rowcol, int chunk) {
    return (chunk ^ ((rowcol >> 1) & 7)) * 8;    // 16B-chunk XOR swizzle
}

__global__ __launch_bounds__(256)
void gemm_glu(const __bf16* __restrict__ Wb,   // [2048][1024]
              const __bf16* __restrict__ Yt,   // [32768][1024]
              const float*  __restrict__ bias, // [2048]
              float* __restrict__ out) {       // [B][H][L]
    __shared__ alignas(16) __bf16 smem[2][16384];   // per buf: Wa 4096 | Wg 4096 | Y 8192

    int t    = threadIdx.x;
    int wave = t >> 5;
    int lane = t & 31;
    int wh = wave >> 2;                 // 0..1
    int wm = wave & 3;                  // 0..3
    int hblk = blockIdx.y * 64;         // 16 blocks over h
    int mblk = blockIdx.x * 128;        // 256 blocks over m
    int r     = lane & 15;
    int khalf = lane >> 4;

    v8f acc_a[2][2] = {{{0}}};
    v8f acc_g[2][2] = {{{0}}};

    // Per-thread fill slots (stage = K=64): Wa/Wg 512 b128 -> 2 each,
    // Y 1024 b128 -> 4 each. Pointers precomputed once, advanced by k0.
    const __bf16* gA[2];
    const __bf16* gY[4];
    int lA[2], lY[4];
    #pragma unroll
    for (int i = 0; i < 2; ++i) {
        int idx = t + i * 256, row = idx >> 3, q = idx & 7;
        gA[i] = Wb + (size_t)(hblk + row) * Hd + q * 8;
        lA[i] = row * 64 + swz(row, q);
    }
    #pragma unroll
    for (int i = 0; i < 4; ++i) {
        int idx = t + i * 256, col = idx >> 3, q = idx & 7;
        gY[i] = Yt + (size_t)(mblk + col) * Hd + q * 8;
        lY[i] = 8192 + col * 64 + swz(col, q);
    }

    // ---- prologue: fill stage 0 into buf 0 ----
    {
        __bf16* sb = smem[0];
        #pragma unroll
        for (int i = 0; i < 2; ++i) {
            cp16(sb + lA[i], gA[i]);                            // W a-half
            cp16(sb + 4096 + lA[i], gA[i] + Hd * Hd);           // W g-half (+2MB imm)
        }
        #pragma unroll
        for (int i = 0; i < 4; ++i) cp16(sb + lY[i], gY[i]);
    }
    async_fence();
    __syncthreads();

    int rowa = wh * 32 + r;             // fragment rows/cols (tile-local)
    int colb = wm * 32 + r;

    for (int kt = 0; kt < 16; ++kt) {   // 16 stages of K=64
        // ---- fill next buffer (free since the kt-1 barrier) ----
        if (kt + 1 < 16) {
            int k0 = (kt + 1) * 64;
            __bf16* sn = smem[(kt + 1) & 1];
            #pragma unroll
            for (int i = 0; i < 2; ++i) {
                cp16(sn + lA[i], gA[i] + k0);
                cp16(sn + 4096 + lA[i], gA[i] + Hd * Hd + k0);
            }
            #pragma unroll
            for (int i = 0; i < 4; ++i) cp16(sn + lY[i], gY[i] + k0);
        }

        // ---- compute from current buffer: 2 K-steps x 8 WMMA ----
        const __bf16* sb = smem[kt & 1];
        #pragma unroll
        for (int ks = 0; ks < 64; ks += 32) {
            int cb = ks >> 3;           // chunk base (0 or 4)
            Frag fa0, fa1, fg0, fg1, fb0, fb1;
            const __bf16* pa0 = sb + rowa * 64;
            const __bf16* pa1 = sb + (rowa + 16) * 64;
            fa0.q[0] = *reinterpret_cast<const uint4*>(pa0 + swz(rowa, cb + khalf));
            fa0.q[1] = *reinterpret_cast<const uint4*>(pa0 + swz(rowa, cb + khalf + 2));
            fa1.q[0] = *reinterpret_cast<const uint4*>(pa1 + swz(rowa + 16, cb + khalf));
            fa1.q[1] = *reinterpret_cast<const uint4*>(pa1 + swz(rowa + 16, cb + khalf + 2));
            fg0.q[0] = *reinterpret_cast<const uint4*>(pa0 + 4096 + swz(rowa, cb + khalf));
            fg0.q[1] = *reinterpret_cast<const uint4*>(pa0 + 4096 + swz(rowa, cb + khalf + 2));
            fg1.q[0] = *reinterpret_cast<const uint4*>(pa1 + 4096 + swz(rowa + 16, cb + khalf));
            fg1.q[1] = *reinterpret_cast<const uint4*>(pa1 + 4096 + swz(rowa + 16, cb + khalf + 2));
            const __bf16* pb0 = sb + 8192 + colb * 64;
            const __bf16* pb1 = sb + 8192 + (colb + 16) * 64;
            int c0 = cb + khalf;
            fb0.q[0] = *reinterpret_cast<const uint4*>(pb0 + swz(colb, c0));
            fb0.q[1] = *reinterpret_cast<const uint4*>(pb0 + swz(colb, c0 + 2));
            fb1.q[0] = *reinterpret_cast<const uint4*>(pb1 + swz(colb + 16, c0));
            fb1.q[1] = *reinterpret_cast<const uint4*>(pb1 + swz(colb + 16, c0 + 2));

            acc_a[0][0] = __builtin_amdgcn_wmma_f32_16x16x32_bf16(false, fa0.v, false, fb0.v, (short)0, acc_a[0][0], false, false);
            acc_a[0][1] = __builtin_amdgcn_wmma_f32_16x16x32_bf16(false, fa0.v, false, fb1.v, (short)0, acc_a[0][1], false, false);
            acc_a[1][0] = __builtin_amdgcn_wmma_f32_16x16x32_bf16(false, fa1.v, false, fb0.v, (short)0, acc_a[1][0], false, false);
            acc_a[1][1] = __builtin_amdgcn_wmma_f32_16x16x32_bf16(false, fa1.v, false, fb1.v, (short)0, acc_a[1][1], false, false);
            acc_g[0][0] = __builtin_amdgcn_wmma_f32_16x16x32_bf16(false, fg0.v, false, fb0.v, (short)0, acc_g[0][0], false, false);
            acc_g[0][1] = __builtin_amdgcn_wmma_f32_16x16x32_bf16(false, fg0.v, false, fb1.v, (short)0, acc_g[0][1], false, false);
            acc_g[1][0] = __builtin_amdgcn_wmma_f32_16x16x32_bf16(false, fg1.v, false, fb0.v, (short)0, acc_g[1][0], false, false);
            acc_g[1][1] = __builtin_amdgcn_wmma_f32_16x16x32_bf16(false, fg1.v, false, fb1.v, (short)0, acc_g[1][1], false, false);
        }

        async_fence();                   // DMA fills of nbuf complete
        __syncthreads();                 // nbuf visible; cbuf free for reuse
    }

    // ---- epilogue: bias + GLU, scatter to out[b][h][l] ----
    // C/D layout: VGPR e holds (M = e + 8*khalf, N = r).
    #pragma unroll
    for (int th = 0; th < 2; ++th) {
        int hb = hblk + wh * 32 + th * 16 + khalf * 8;
        #pragma unroll
        for (int tm = 0; tm < 2; ++tm) {
            int m = mblk + wm * 32 + tm * 16 + r;
            int bidx = m >> 12;             // L == 4096
            int l    = m & (Ld - 1);
            #pragma unroll
            for (int e = 0; e < 8; ++e) {
                int hh = hb + e;
                float av = acc_a[th][tm][e] + bias[hh];
                float gv = acc_g[th][tm][e] + bias[Hd + hh];
                float o  = av * (1.0f / (1.0f + expf(-gv)));
                __builtin_nontemporal_store(o, out + ((size_t)bidx * Hd + hh) * Ld + l);
            }
        }
    }
}

// ---------------------------------------------------------------------------
extern "C" void kernel_launch(void* const* d_in, const int* in_sizes, int n_in,
                              void* d_out, int out_size, void* d_ws, size_t ws_size,
                              hipStream_t stream) {
    const float* u        = (const float*)d_in[0];
    const float* log_dt   = (const float*)d_in[1];
    const float* C_real   = (const float*)d_in[2];
    const float* logAreal = (const float*)d_in[3];
    const float* A_imag   = (const float*)d_in[4];
    const float* Dv       = (const float*)d_in[5];
    const float* W_out    = (const float*)d_in[6];
    const float* b_out    = (const float*)d_in[7];
    float* out = (float*)d_out;

    char* ws = (char*)d_ws;
    float*          cst = (float*)ws;                                      // 512 KB
    __hip_bfloat16* Wb  = (__hip_bfloat16*)(ws + (512 << 10));             // 4 MB
    __hip_bfloat16* Yt  = (__hip_bfloat16*)(ws + (512 << 10) + (4 << 20)); // 64 MB

    ssm_prep_consts<<<(Hd * N2d) / 256, 256, 0, stream>>>(log_dt, C_real, logAreal, A_imag, cst);
    w_to_bf16<<<(2 * Hd * Hd) / 256, 256, 0, stream>>>(W_out, Wb);
    ssm_scan<<<(Bd * Hd / 4) / 8, 256, 0, stream>>>(u, cst, Dv, Yt);
    dim3 grid(Md / 128, Hd / 64);
    gemm_glu<<<grid, 256, 0, stream>>>((const __bf16*)Wb, (const __bf16*)Yt, b_out, out);
}
